// DynamicGraphConvLayer_27882927685999
// MI455X (gfx1250) — matrix operand order repaired and checked by reading, not compiled, not run on hardware
//
#include <hip/hip_runtime.h>
#include <cmath>

typedef __attribute__((ext_vector_type(16))) _Float16 v16h;
typedef __attribute__((ext_vector_type(2)))  __fp16   v2fp;
typedef __attribute__((ext_vector_type(8)))  float    v8f;

#define NKEYS 4096
#define FDIM  64
#define BATCH 4

union V16HU { v16h v; uint4 q[2]; };
union PBU   { v16h v; unsigned int u[8]; };
union H2U   { v2fp h; unsigned int u; };

// Load an A/B fragment as two 16-byte contiguous pieces (low 8 halfs -> v0..3,
// high 8 halfs -> v4..7), matching the CDNA5 16-bit matrix VGPR layouts.
__device__ __forceinline__ v16h load16(const _Float16* p0, const _Float16* p1) {
    V16HU u;
    u.q[0] = *reinterpret_cast<const uint4*>(p0);
    u.q[1] = *reinterpret_cast<const uint4*>(p1);
    return u.v;
}

__device__ __forceinline__ v8f wmma_f16(v16h a, v16h b, v8f c) {
    return __builtin_amdgcn_wmma_f32_16x16x32_f16(false, a, false, b, (short)0, c,
                                                  false, false);
}

// xor-16 half-wave swap: single v_permlanex16_b32 (identity selects) instead of
// ds_bpermute + s_wait_dscnt.
__device__ __forceinline__ unsigned int xor16_u(unsigned int x) {
#if __has_builtin(__builtin_amdgcn_permlanex16)
    return __builtin_amdgcn_permlanex16(x, x, 0x76543210u, 0xfedcba98u, true, false);
#else
    return __shfl_xor((int)x, 16, 32);
#endif
}
__device__ __forceinline__ float xor16_f(float x) {
    return __uint_as_float(xor16_u(__float_as_uint(x)));
}

__device__ __forceinline__ unsigned int pack2(float x, float y) {
    H2U t;
    t.h = __builtin_amdgcn_cvt_pkrtz(x, y);   // v_cvt_pk_rtz_f16_f32
    return t.u;
}

// ---------------------------------------------------------------------------
// Kernel 1: proj = relu(x @ Wp + bp) -> f16 [B][N][64], plus a log2(e)-scaled
// copy used as the Q operand (lets the softmax run on native base-2 exp).
// ---------------------------------------------------------------------------
__global__ void __launch_bounds__(256)
proj_kernel(const float* __restrict__ x, const float* __restrict__ Wp,
            const float* __restrict__ bp, _Float16* __restrict__ proj,
            _Float16* __restrict__ projQ) {
    int idx = blockIdx.x * 256 + threadIdx.x;   // over B*N*64
    int d = idx & 63;
    int n = idx >> 6;                           // row in [0, B*N)
    const float* xr = x + (size_t)n * FDIM;
    float acc = bp[d];
#pragma unroll 8
    for (int f = 0; f < FDIM; ++f) acc = fmaf(xr[f], Wp[f * FDIM + d], acc);
    float r = fmaxf(acc, 0.0f);
    proj[idx]  = (_Float16)r;
    projQ[idx] = (_Float16)(r * 1.44269504088896340736f);   // * log2(e)
}

// ---------------------------------------------------------------------------
// Kernel 2: xT[b][f][n] = (f16) x[b][n][f]   (V^T, row-major over keys)
// ---------------------------------------------------------------------------
__global__ void __launch_bounds__(256)
transpose_kernel(const float* __restrict__ x, _Float16* __restrict__ xT) {
    int idx = blockIdx.x * 256 + threadIdx.x;   // over B*64*N
    int n = idx & (NKEYS - 1);
    int f = (idx >> 12) & 63;
    int b = idx >> 18;
    xT[idx] = (_Float16)x[((size_t)b * NKEYS + n) * FDIM + f];
}

// ---------------------------------------------------------------------------
// Kernel 3: fused flash attention:  H = softmax(proj @ proj^T) @ x
//   One wave per 16-query tile. Computes S^T = K@Q^T so per-query softmax
//   reductions are in-lane + one permlanex16. PV done as H^T = V^T @ P^T.
//   All softmax math in log2 domain (Q pre-scaled by log2 e).
// ---------------------------------------------------------------------------
__global__ void __launch_bounds__(128)
attn_kernel(const _Float16* __restrict__ proj,    // [B][N][64] (K operand)
            const _Float16* __restrict__ projQ,   // [B][N][64] (Q, *log2e)
            const _Float16* __restrict__ xT,      // [B][64][N] (V^T)
            float* __restrict__ H) {              // [B][N][64]
    const int nqb  = NKEYS / 64;
    const int b    = blockIdx.x / nqb;
    const int qblk = blockIdx.x % nqb;
    const int w    = threadIdx.x >> 5;
    const int lane = threadIdx.x & 31;
    const int lq   = lane & 15;
    const int hi   = lane >> 4;
    const int q0   = qblk * 64 + w * 16;

    const _Float16* projb = proj  + (size_t)b * NKEYS * FDIM;
    const _Float16* qb    = projQ + (size_t)b * NKEYS * FDIM;
    const _Float16* xTb   = xT    + (size_t)b * FDIM * NKEYS;
    float*          Hb    = H     + (size_t)b * NKEYS * FDIM;

    // Q^T as B-fragments (fixed for whole key loop):
    // lanes 0-15 -> K(features) 0..7,16..23, lanes 16-31 -> 8..15,24..31 (then +32)
    const _Float16* qrow = qb + (size_t)(q0 + lq) * FDIM;
    v16h qb0 = load16(qrow + hi * 8, qrow + 16 + hi * 8);
    v16h qb1 = load16(qrow + 32 + hi * 8, qrow + 48 + hi * 8);

    v8f o0 = {}, o1 = {}, o2 = {}, o3 = {};      // H^T accumulators
    float m = -INFINITY;                          // log2-domain running max
    float l = 0.0f;

    for (int jk = 0; jk < NKEYS; jk += 32) {
        // ---- S^T = K_tile(32x64) @ Q^T(64x16), two 16-key C fragments ----
        const _Float16* kr0 = projb + (size_t)(jk + lq) * FDIM + hi * 8;
        const _Float16* kr1 = projb + (size_t)(jk + 16 + lq) * FDIM + hi * 8;
        const _Float16* vr  = xTb + (size_t)lq * NKEYS + jk + hi * 8;
        const size_t vs = (size_t)16 * NKEYS;

        // prefetch two tiles ahead (global_prefetch_b8)
        __builtin_prefetch(kr0 + 64 * FDIM, 0, 1);
        __builtin_prefetch(vr + 64, 0, 1);

        v8f s0 = {}, s1 = {};
        s0 = wmma_f16(load16(kr0, kr0 + 16), qb0, s0);        // feats 0..31
        s0 = wmma_f16(load16(kr0 + 32, kr0 + 48), qb1, s0);   // feats 32..63
        s1 = wmma_f16(load16(kr1, kr1 + 16), qb0, s1);
        s1 = wmma_f16(load16(kr1 + 32, kr1 + 48), qb1, s1);

        // ---- online softmax (log2 domain, per query = per fragment col) ----
        float tmax = s0[0];
#pragma unroll
        for (int v = 1; v < 8; ++v) tmax = fmaxf(tmax, s0[v]);
#pragma unroll
        for (int v = 0; v < 8; ++v) tmax = fmaxf(tmax, s1[v]);
        tmax = fmaxf(tmax, xor16_f(tmax));
        float mn   = fmaxf(m, tmax);
        float corr = __builtin_amdgcn_exp2f(m - mn);
        float p0[8], p1[8];
        float ts = 0.0f;
#pragma unroll
        for (int v = 0; v < 8; ++v) { p0[v] = __builtin_amdgcn_exp2f(s0[v] - mn); ts += p0[v]; }
#pragma unroll
        for (int v = 0; v < 8; ++v) { p1[v] = __builtin_amdgcn_exp2f(s1[v] - mn); ts += p1[v]; }
        ts += xor16_f(ts);
        l = l * corr + ts;
        m = mn;
#pragma unroll
        for (int v = 0; v < 8; ++v) {
            o0[v] *= corr; o1[v] *= corr; o2[v] *= corr; o3[v] *= corr;
        }

        // ---- rebuild P^T as a B-fragment (K = 32 keys, N = 16 queries) ----
        // Pack pairs to f16x2; each lane already holds its own keys, so pb takes
        // the dwords unswapped (sa/sc unused): 8 cvt_pk.
        PBU pb;
#pragma unroll
        for (int j = 0; j < 4; ++j) {
            unsigned int a  = pack2(p0[2 * j], p0[2 * j + 1]);
            unsigned int c  = pack2(p1[2 * j], p1[2 * j + 1]);
            unsigned int sa = xor16_u(a);
            unsigned int sc = xor16_u(c);
            pb.u[j]     = a;    // keys hi*8 + 2j,2j+1
            pb.u[j + 4] = c;    // keys 16 + hi*8 + 2j,2j+1
        }

        // ---- O^T += V^T(64x32) @ P^T(32x16), 4 feature chunks of 16 ----
        o0 = wmma_f16(load16(vr,          vr + 16),          pb.v, o0);
        o1 = wmma_f16(load16(vr + vs,     vr + vs + 16),     pb.v, o1);
        o2 = wmma_f16(load16(vr + 2 * vs, vr + 2 * vs + 16), pb.v, o2);
        o3 = wmma_f16(load16(vr + 3 * vs, vr + 3 * vs + 16), pb.v, o3);
    }

    // ---- epilogue: H[q][feat] = O^T[feat][q] / l ----
    float inv_l = 1.0f / l;
    float* hout = Hb + (size_t)(q0 + lq) * FDIM + hi * 8;
#pragma unroll
    for (int v = 0; v < 8; ++v) {
        hout[v]      = o0[v] * inv_l;
        hout[16 + v] = o1[v] * inv_l;
        hout[32 + v] = o2[v] * inv_l;
        hout[48 + v] = o3[v] * inv_l;
    }
}

// ---------------------------------------------------------------------------
// Kernel 4: out = tanh(elu(H @ W1 + b1) @ W2 + b2)   (4 rows per block)
// ---------------------------------------------------------------------------
__global__ void __launch_bounds__(256)
mlp_kernel(const float* __restrict__ H, const float* __restrict__ W1,
           const float* __restrict__ b1, const float* __restrict__ W2,
           const float* __restrict__ b2, float* __restrict__ out) {
    __shared__ float sh[4][64];
    int r = threadIdx.x >> 6;
    int e = threadIdx.x & 63;
    size_t row = (size_t)blockIdx.x * 4 + r;
    const float* hrow = H + row * FDIM;
    float acc = b1[e];
#pragma unroll 8
    for (int d2 = 0; d2 < FDIM; ++d2) acc = fmaf(hrow[d2], W1[d2 * FDIM + e], acc);
    sh[r][e] = acc > 0.0f ? acc : (__expf(acc) - 1.0f);   // ELU
    __syncthreads();
    float acc2 = b2[e];
#pragma unroll 8
    for (int d2 = 0; d2 < FDIM; ++d2) acc2 = fmaf(sh[r][d2], W2[d2 * FDIM + e], acc2);
    out[row * FDIM + e] = tanhf(acc2);
}

// ---------------------------------------------------------------------------
extern "C" void kernel_launch(void* const* d_in, const int* in_sizes, int n_in,
                              void* d_out, int out_size, void* d_ws, size_t ws_size,
                              hipStream_t stream) {
    const float* x  = (const float*)d_in[0];
    const float* Wp = (const float*)d_in[1];
    const float* bp = (const float*)d_in[2];
    const float* W1 = (const float*)d_in[3];
    const float* b1 = (const float*)d_in[4];
    const float* W2 = (const float*)d_in[5];
    const float* b2 = (const float*)d_in[6];
    float* out = (float*)d_out;

    char* ws = (char*)d_ws;
    _Float16* proj  = (_Float16*)ws;                               // 2 MB
    _Float16* projQ = (_Float16*)(ws + (size_t)2 * 1024 * 1024);   // 2 MB
    _Float16* xT    = (_Float16*)(ws + (size_t)4 * 1024 * 1024);   // 2 MB
    float*    H     = (float*)   (ws + (size_t)6 * 1024 * 1024);   // 4 MB

    const int total = BATCH * NKEYS * FDIM;                        // 1,048,576
    proj_kernel     <<<total / 256, 256, 0, stream>>>(x, Wp, bp, proj, projQ);
    transpose_kernel<<<total / 256, 256, 0, stream>>>(x, xT);
    attn_kernel     <<<BATCH * (NKEYS / 64), 128, 0, stream>>>(proj, projQ, xT, H);
    mlp_kernel      <<<BATCH * NKEYS / 4, 256, 0, stream>>>(H, W1, b1, W2, b2, out);
}
